// ChildSumTreeLSTMOp_63385127354391
// MI455X (gfx1250) — matrix operand order repaired and checked by reading, not compiled, not run on hardware
//
#include <hip/hip_runtime.h>
#include <hip/hip_bf16.h>
#include <stdint.h>

// ---------------------------------------------------------------------------
// Child-Sum Tree-LSTM on CDNA5 (gfx1250), wave32, v_wmma_f32_16x16x32_bf16.
// N=65536 nodes, DIM=128, L=8 levels (depth sorted ascending).
// ---------------------------------------------------------------------------

typedef __bf16 bf16_t;
typedef __attribute__((ext_vector_type(16))) __bf16 bf16x16;
typedef __attribute__((ext_vector_type(8)))  float  floatx8;

#define DIM   128
#define LVLS  8
#define APAD  136   // LDS row stride (bf16 elems) to dodge bank conflicts

__device__ __forceinline__ unsigned short f2bf_bits(float f) {
    unsigned u = __float_as_uint(f);
    unsigned r = u + 0x7FFFu + ((u >> 16) & 1u);   // round-to-nearest-even
    return (unsigned short)(r >> 16);
}
__device__ __forceinline__ bf16_t f2bf(float f) {
    union { unsigned short u; bf16_t b; } cv; cv.u = f2bf_bits(f); return cv.b;
}
__device__ __forceinline__ float sigf(float x) {
    return 1.0f / (1.0f + __expf(-x));
}
__device__ __forceinline__ float tanh_f(float x) {
    float e = __expf(2.0f * x);                    // e in [0, inf]
    return 1.0f - 2.0f / (e + 1.0f);               // robust at +-inf
}

// A-fragment (16x32 bf16, MxK) gather from LDS in this code's contiguous k order (matches B pack):
// lanes 0-15: M=lane, K = kBase + {0..15}; lanes 16-31: K += 16.
__device__ __forceinline__ bf16x16
load_a_frag(const unsigned short (*aStage)[APAD], int lane, int kBase) {
    int M  = lane & 15;
    int ko = kBase + ((lane & 16) ? 16 : 0);
    const bf16_t* p = (const bf16_t*)&aStage[M][ko];
    bf16x16 a;
#pragma unroll
    for (int e = 0; e < 8; ++e) { a[e] = p[e]; a[8 + e] = p[8 + e]; }
    return a;
}

// B fragments are pre-packed so each lane reads 16 contiguous bf16 (32B).
__device__ __forceinline__ bf16x16
load_b_frag(const unsigned short* pack, int fragId, int lane) {
    return *(const bf16x16*)(pack + (size_t)(fragId * 32 + lane) * 16);
}

__device__ __forceinline__ floatx8
wmma_bf16(bf16x16 a, bf16x16 b, floatx8 c) {
    return __builtin_amdgcn_wmma_f32_16x16x32_bf16(false, a, false, b,
                                                   (short)0, c, false, false);
}

// One 16x16 C tile: A from LDS, B from packed weights, K=128 -> 4 WMMAs.
__device__ __forceinline__ floatx8
mm_tile(const unsigned short (*aStage)[APAD], const unsigned short* pack,
        int nt, int lane) {
    floatx8 acc = {};
#pragma unroll
    for (int kt = 0; kt < 4; ++kt) {
        bf16x16 a = load_a_frag(aStage, lane, kt * 32);
        bf16x16 b = load_b_frag(pack, nt * 4 + kt, lane);
        acc = wmma_bf16(a, b, acc);
    }
    return acc;
}

// C-tile store with compile-time row stride -> immediate-offset stores.
template <int LDO>
__device__ __forceinline__ void
store_tile(float* __restrict__ out, int row0, int col, floatx8 acc) {
    float* p = out + (size_t)row0 * LDO + col;
#pragma unroll
    for (int r = 0; r < 8; ++r) p[r * LDO] = acc[r];
}

// ---------------------------------------------------------------------------
// Zero h_sum, fc_sum, n_child once per launch.
__global__ void init_zero_kernel(float* hsum, float* fcsum, int* nchild,
                                 size_t n128, size_t n) {
    size_t stride = (size_t)gridDim.x * blockDim.x;
    for (size_t i = blockIdx.x * (size_t)blockDim.x + threadIdx.x;
         i < 2 * n128 + n; i += stride) {
        if (i < n128)           hsum[i] = 0.0f;
        else if (i < 2 * n128)  fcsum[i - n128] = 0.0f;
        else                    nchild[i - 2 * n128] = 0;
    }
}

// ---------------------------------------------------------------------------
// Pack f32 weights into bf16 B-fragment-major layout.
// B-fragment (32x16, KxN): lane l -> N = l&15, K = kBase + (l>=16?16:0) + e.
// Fragment id = nt*4 + kt. Blocks: 0..95 Wioux, 96..127 Wfx,
// 128..223 Wiouh, 224..255 Wfh.
__global__ void packw_kernel(const float* Wioux, const float* Wiouh,
                             const float* Wfx,   const float* Wfh,
                             unsigned short* pIoux, unsigned short* pIouh,
                             unsigned short* pFx,   unsigned short* pFh) {
    int b = blockIdx.x, lane = threadIdx.x;
    const float* W; unsigned short* P; int f, ld;
    if (b < 96)       { W = Wioux; P = pIoux; f = b;       ld = 384; }
    else if (b < 128) { W = Wfx;   P = pFx;   f = b - 96;  ld = 128; }
    else if (b < 224) { W = Wiouh; P = pIouh; f = b - 128; ld = 384; }
    else              { W = Wfh;   P = pFh;   f = b - 224; ld = 128; }
    int nt = f >> 2, kt = f & 3;
    int Nc = nt * 16 + (lane & 15);
    int k0 = kt * 32 + ((lane & 16) ? 16 : 0);
    bf16x16 v;
#pragma unroll
    for (int e = 0; e < 16; ++e) v[e] = f2bf(W[(size_t)(k0 + e) * ld + Nc]);
    *(bf16x16*)(P + (size_t)(f * 32 + lane) * 16) = v;
}

// ---------------------------------------------------------------------------
// xi = x @ Wioux (Nx384), xf = x @ Wfx (Nx128), fused. 16-row tile per block,
// 8 waves x 4 col-tiles = 32 col-tiles (24 xi + 8 xf), K=128 -> 4 WMMA each.
__global__ void gemmx_kernel(const float* __restrict__ x,
                             const unsigned short* __restrict__ pIoux,
                             const unsigned short* __restrict__ pFx,
                             float* __restrict__ xi, float* __restrict__ xf) {
    __shared__ unsigned short aStage[16][APAD];
    int tid = threadIdx.x, j0 = blockIdx.x * 16;
#pragma unroll
    for (int q = 0; q < 8; ++q) {
        int idx = tid + q * 256, M = idx >> 7, k = idx & 127;
        aStage[M][k] = f2bf_bits(x[(size_t)(j0 + M) * DIM + k]);
    }
    __syncthreads();
    int wave = tid >> 5, lane = tid & 31;
    int rOff = (lane & 16) ? 8 : 0, nc = lane & 15;
#pragma unroll
    for (int q = 0; q < 4; ++q) {
        int ct = wave * 4 + q;                    // uniform per wave
        if (ct < 24) {
            floatx8 acc = mm_tile(aStage, pIoux, ct, lane);
            store_tile<384>(xi, j0 + rOff, ct * 16 + nc, acc);
        } else {
            floatx8 acc = mm_tile(aStage, pFx, ct - 24, lane);
            store_tile<128>(xf, j0 + rOff, (ct - 24) * 16 + nc, acc);
        }
    }
}

// ---------------------------------------------------------------------------
// Child pass, level d: for rows j with depth[j]==d+1:
//   f = sigmoid(xf[parent] + h_j @ Wfh); atomic: fc_sum[par]+=f*c_j,
//   h_sum[par]+=h_j, n_child[par]+=1.
__global__ void fchild_kernel(const float* __restrict__ h,
                              const float* __restrict__ cbuf,
                              const float* __restrict__ xf,
                              const int* __restrict__ depth,
                              const int* __restrict__ parent,
                              const unsigned short* __restrict__ pFh,
                              float* __restrict__ hsum,
                              float* __restrict__ fcsum,
                              int* __restrict__ nchild, int d) {
    __shared__ unsigned short aStage[16][APAD];
    __shared__ int parS[16], actS[16], flag;
    int tid = threadIdx.x, j0 = blockIdx.x * 16;
    if (tid == 0) flag = 0;
    __syncthreads();
    if (tid < 16) {
        int act = (depth[j0 + tid] == d + 1) ? 1 : 0;
        actS[tid] = act;
        parS[tid] = parent[j0 + tid];
        if (act) { atomicOr(&flag, 1); atomicAdd(&nchild[parS[tid]], 1); }
    }
    __syncthreads();
    if (!flag) return;
#pragma unroll
    for (int q = 0; q < 16; ++q) {        // stage h (f32 -> bf16) into LDS
        int idx = tid + q * 128, M = idx >> 7, k = idx & 127;
        aStage[M][k] = f2bf_bits(h[(size_t)(j0 + M) * DIM + k]);
    }
    __syncthreads();
    int wave = tid >> 5, lane = tid & 31;
    int rOff = (lane & 16) ? 8 : 0, nc = lane & 15;
#pragma unroll
    for (int q = 0; q < 2; ++q) {
        int nt = wave * 2 + q;
        floatx8 acc = mm_tile(aStage, pFh, nt, lane);
#pragma unroll
        for (int r = 0; r < 8; ++r) {
            int M = r + rOff;
            if (!actS[M]) continue;
            int row = j0 + M, col = nt * 16 + nc, p = parS[M];
            float fv = sigf(acc[r] + xf[(size_t)p * DIM + col]);
            float cv = cbuf[(size_t)row * DIM + col];
            atomicAdd(&fcsum[(size_t)p * DIM + col], fv * cv);
            atomicAdd(&hsum [(size_t)p * DIM + col],
                      h[(size_t)row * DIM + col]);
        }
    }
}

// ---------------------------------------------------------------------------
// Update pass, level d: rows p with depth[p]==d.
// leaf (n_child==0): h=x, c=tanh(x). else:
//   [i,o,u] = xi[p] + h_sum[p] @ Wiouh; c = fc_sum + sig(i)*tanh(u);
//   h = sig(o)*tanh(c).
__global__ void upd_kernel(const float* __restrict__ x,
                           const float* __restrict__ xi,
                           const float* __restrict__ hsum,
                           const float* __restrict__ fcsum,
                           const int* __restrict__ nchild,
                           const int* __restrict__ depth,
                           const unsigned short* __restrict__ pIouh,
                           float* __restrict__ h, float* __restrict__ cbuf,
                           int d) {
    __shared__ unsigned short aStage[16][APAD];
    __shared__ float iouS[16][384];
    __shared__ int actS[16], nchS[16], flag;
    int tid = threadIdx.x, j0 = blockIdx.x * 16;
    if (tid == 0) flag = 0;
    __syncthreads();
    if (tid < 16) {
        int act = (depth[j0 + tid] == d) ? 1 : 0;
        actS[tid] = act;
        nchS[tid] = nchild[j0 + tid];
        if (act) atomicOr(&flag, 1);
    }
    __syncthreads();
    if (!flag) return;
#pragma unroll
    for (int q = 0; q < 8; ++q) {         // stage h_sum (f32 -> bf16)
        int idx = tid + q * 256, M = idx >> 7, k = idx & 127;
        aStage[M][k] = f2bf_bits(hsum[(size_t)(j0 + M) * DIM + k]);
    }
    __syncthreads();
    int wave = tid >> 5, lane = tid & 31;
    int rOff = (lane & 16) ? 8 : 0, nc = lane & 15;
#pragma unroll
    for (int q = 0; q < 3; ++q) {         // 8 waves x 3 = 24 col-tiles (384)
        int nt = wave * 3 + q;
        floatx8 acc = mm_tile(aStage, pIouh, nt, lane);
        store_tile<384>(&iouS[0][0], rOff, nt * 16 + nc, acc);
    }
    __syncthreads();
#pragma unroll
    for (int q = 0; q < 8; ++q) {         // gates, 16 rows x 128 cols
        int idx = tid + q * 256, M = idx >> 7, col = idx & 127;
        if (!actS[M]) continue;
        int row = j0 + M;
        if (nchS[M] == 0) {               // leaf
            float xv = x[(size_t)row * DIM + col];
            h[(size_t)row * DIM + col]    = xv;
            cbuf[(size_t)row * DIM + col] = tanh_f(xv);
        } else {
            const float* xr = xi + (size_t)row * 384 + col;
            float i_ = xr[0]   + iouS[M][col];
            float o_ = xr[128] + iouS[M][128 + col];
            float u_ = xr[256] + iouS[M][256 + col];
            float cv = fcsum[(size_t)row * DIM + col]
                       + sigf(i_) * tanh_f(u_);
            h[(size_t)row * DIM + col]    = sigf(o_) * tanh_f(cv);
            cbuf[(size_t)row * DIM + col] = cv;
        }
    }
}

// ---------------------------------------------------------------------------
extern "C" void kernel_launch(void* const* d_in, const int* in_sizes, int n_in,
                              void* d_out, int out_size, void* d_ws,
                              size_t ws_size, hipStream_t stream) {
    const float* x      = (const float*)d_in[0];
    const int*   parent = (const int*)  d_in[1];
    const int*   depth  = (const int*)  d_in[2];
    const float* Wioux  = (const float*)d_in[3];
    const float* Wiouh  = (const float*)d_in[4];
    const float* Wfx    = (const float*)d_in[5];
    const float* Wfh    = (const float*)d_in[6];
    const int N = in_sizes[0] / DIM;     // 65536

    // workspace carve-out (all 256B aligned)
    char* ws = (char*)d_ws;
    size_t off = 0;
    auto carve = [&](size_t bytes) {
        char* p = ws + off;
        off = (off + bytes + 255) & ~(size_t)255;
        return p;
    };
    float* xi    = (float*)carve((size_t)N * 384 * 4);
    float* xf    = (float*)carve((size_t)N * DIM * 4);
    float* cbuf  = (float*)carve((size_t)N * DIM * 4);
    float* hsum  = (float*)carve((size_t)N * DIM * 4);
    float* fcsum = (float*)carve((size_t)N * DIM * 4);
    int*   nchd  = (int*)  carve((size_t)N * 4);
    unsigned short* pIoux = (unsigned short*)carve((size_t)DIM * 384 * 2);
    unsigned short* pIouh = (unsigned short*)carve((size_t)DIM * 384 * 2);
    unsigned short* pFx   = (unsigned short*)carve((size_t)DIM * DIM * 2);
    unsigned short* pFh   = (unsigned short*)carve((size_t)DIM * DIM * 2);

    float* h = (float*)d_out;            // h lives directly in the output

    init_zero_kernel<<<2048, 256, 0, stream>>>(hsum, fcsum, nchd,
                                               (size_t)N * DIM, (size_t)N);
    packw_kernel<<<256, 32, 0, stream>>>(Wioux, Wiouh, Wfx, Wfh,
                                         pIoux, pIouh, pFx, pFh);
    gemmx_kernel<<<N / 16, 256, 0, stream>>>(x, pIoux, pFx, xi, xf);

    for (int d = LVLS - 1; d >= 0; --d) {
        fchild_kernel<<<N / 16, 128, 0, stream>>>(h, cbuf, xf, depth, parent,
                                                  pFh, hsum, fcsum, nchd, d);
        upd_kernel<<<N / 16, 256, 0, stream>>>(x, xi, hsum, fcsum, nchd,
                                               depth, pIouh, h, cbuf, d);
    }
    (void)n_in; (void)out_size; (void)ws_size; (void)in_sizes;
}
